// VerticalRTColumnNet_13417477833261
// MI455X (gfx1250) — matrix operand order repaired and checked by reading, not compiled, not run on hardware
//
#include <hip/hip_runtime.h>
#include <hip/hip_bf16.h>

typedef _Float16 v16h __attribute__((ext_vector_type(16)));
typedef _Float16 v8h  __attribute__((ext_vector_type(8)));
typedef float    v8f  __attribute__((ext_vector_type(8)));

// f16 workspace layout (offsets in halves)
#define WS_ENC1 0        // [64][32]  enc_w1^T (k padded 22->32)
#define WS_ENC2 2048     // [64][64]  enc_w2^T
#define WS_GW1  6144     // 4 x [64][64]  gate w1^T (td,cd,tu,cu)
#define WS_GW2  22528    // 4 x [16][64]  gate w2^T (n padded 4->16)
#define WS_SDS  26624    // [16][64]  cols: sd(0..3) su(4..7) skip(8..11) pad

#define WMMA(a,b,c) __builtin_amdgcn_wmma_f32_16x16x32_f16(false,(a),false,(b),(short)0,(c),false,false)

// -------- prepack: f32 weights -> f16 transposed B-operand layout ----------
__global__ void prepack_kernel(const float* ew1, const float* ew2,
                               const float* tdw1, const float* cdw1,
                               const float* tuw1, const float* cuw1,
                               const float* tdw2, const float* cdw2,
                               const float* tuw2, const float* cuw2,
                               const float* sdw, const float* suw,
                               const float* skw, _Float16* ws) {
  int tid = blockIdx.x * blockDim.x + threadIdx.x;
  int nth = gridDim.x * blockDim.x;
  for (int i = tid; i < 2048; i += nth) {            // enc1^T [n][k], K=32
    int n = i >> 5, k = i & 31;
    ws[WS_ENC1 + i] = (_Float16)((k < 22) ? ew1[k * 64 + n] : 0.f);
  }
  for (int i = tid; i < 4096; i += nth) {            // enc2^T [n][k], K=64
    int n = i >> 6, k = i & 63;
    ws[WS_ENC2 + i] = (_Float16)ew2[k * 64 + n];
  }
  const float* g1[4] = {tdw1, cdw1, tuw1, cuw1};
  for (int i = tid; i < 16384; i += nth) {           // gate w1^T
    int g = i >> 12, r = i & 4095, n = r >> 6, k = r & 63;
    ws[WS_GW1 + i] = (_Float16)g1[g][k * 64 + n];
  }
  const float* g2[4] = {tdw2, cdw2, tuw2, cuw2};
  for (int i = tid; i < 4096; i += nth) {            // gate w2^T, n padded
    int g = i >> 10, r = i & 1023, n = r >> 6, k = r & 63;
    ws[WS_GW2 + i] = (_Float16)((n < 4) ? g2[g][k * 4 + n] : 0.f);
  }
  for (int i = tid; i < 1024; i += nth) {            // [sd|su|skip]^T fused
    int n = i >> 6, k = i & 63;
    float v = (n < 4) ? sdw[k * 4 + n]
            : (n < 8) ? suw[k * 4 + (n - 4)]
            : (n < 12) ? skw[k * 4 + (n - 8)] : 0.f;
    ws[WS_SDS + i] = (_Float16)v;
  }
}

// ---- fragment loaders per CDNA5 16-bit WMMA VGPR layouts (wave32) ---------
// A (16x32): lane m=lane&15; elem i -> k = (lane>>4)*8 + (i<8 ? i : i+8)
__device__ __forceinline__ v16h ldsA(const _Float16* tb, int lane, int kOff) {
  int m = lane & 15;
  int ks = (lane >> 4) * 8;
  const _Float16* p = tb + m * 64 + kOff + ks;
  v8h lo = *(const v8h*)(p);
  v8h hi = *(const v8h*)(p + 16);
  return __builtin_shufflevector(lo, hi, 0,1,2,3,4,5,6,7,8,9,10,11,12,13,14,15);
}
// B (32x16): lane n=lane&15; elem i -> k = kt*32 + (lane>>4)*8 + (i<8 ? i : i+8)
__device__ __forceinline__ v16h glbB(const _Float16* wt, int lane, int K,
                                     int nt, int kt) {
  int n = nt * 16 + (lane & 15);
  int kb = kt * 32 + (lane >> 4) * 8;
  const v8h* p = (const v8h*)(wt + n * K + kb);
  v8h lo = p[0], hi = p[2];
  return __builtin_shufflevector(lo, hi, 0,1,2,3,4,5,6,7,8,9,10,11,12,13,14,15);
}
// C (16x16 f32): lane n=lane&15; vgpr v -> m = v + (lane>>4)*8
__device__ __forceinline__ void storeReluTile(_Float16* tile, int lane, int nt,
                                              const float* bias, v8f c) {
  int n = lane & 15, mb = (lane >> 4) * 8;
  float bs = bias[nt * 16 + n];
#pragma unroll
  for (int v = 0; v < 8; ++v) {
    float xv = c[v] + bs;
    xv = xv > 0.f ? xv : 0.f;
    tile[(mb + v) * 64 + nt * 16 + n] = (_Float16)xv;
  }
}

// ------------------------------ fused kernel -------------------------------
__global__ __launch_bounds__(320)
void vrtcn_fused(const float* __restrict__ x, const float* __restrict__ emb,
                 const float* __restrict__ enc_b1, const float* __restrict__ enc_b2,
                 const float* __restrict__ td_b1, const float* __restrict__ td_b2,
                 const float* __restrict__ cd_b1, const float* __restrict__ cd_b2,
                 const float* __restrict__ tu_b1, const float* __restrict__ tu_b2,
                 const float* __restrict__ cu_b1, const float* __restrict__ cu_b2,
                 const float* __restrict__ sd_b, const float* __restrict__ su_b,
                 const float* __restrict__ skip_b,
                 const float* __restrict__ bc_w1, const float* __restrict__ bc_b1,
                 const float* __restrict__ bc_w2, const float* __restrict__ bc_b2,
                 const float* __restrict__ hd_w1, const float* __restrict__ hd_b1,
                 const float* __restrict__ hd_w2, const float* __restrict__ hd_b2,
                 const float* __restrict__ hu_w1, const float* __restrict__ hu_b1,
                 const float* __restrict__ hu_w2, const float* __restrict__ hu_b2,
                 const _Float16* __restrict__ ws, float* __restrict__ out, int Bn) {
  const int w = threadIdx.x >> 5;    // wave id == layer index (0..9)
  const int lane = threadIdx.x & 31;
  const int b0 = blockIdx.x * 16;

  __shared__ _Float16 bufA[10][1024];   // per-wave 16x64 f16 tile
  __shared__ _Float16 bufB[10][1024];   // per-wave 16x64 f16 tile
  __shared__ float stok[160][36];       // per token: Td Cd Tu Cu Sd Su Sk D U

  _Float16* A  = bufA[w];
  _Float16* Bt = bufB[w];

  // ---- stage xe (16 tokens x 32 features, f16, k>=22 zero) ----
  if (lane < 16) {
    int bb = lane;
    int bs = b0 + bb; if (bs >= Bn) bs = Bn - 1;
    const float* xr = x + (size_t)bs * 60;   // x is (B, C=6, L=10)
#pragma unroll
    for (int c = 0; c < 6; ++c) A[bb * 64 + c] = (_Float16)xr[c * 10 + w];
#pragma unroll
    for (int e = 0; e < 16; ++e) A[bb * 64 + 6 + e] = (_Float16)emb[w * 16 + e];
#pragma unroll
    for (int k = 22; k < 32; ++k) A[bb * 64 + k] = (_Float16)0.f;
  }

  // ---- enc1: h1 = relu(xe @ enc_w1 + b1) ----
  {
    v16h a = ldsA(A, lane, 0);
#pragma unroll
    for (int nt = 0; nt < 4; ++nt) {
      v8f c = {};
      c = WMMA(a, glbB(ws + WS_ENC1, lane, 32, nt, 0), c);
      storeReluTile(Bt, lane, nt, enc_b1, c);
    }
  }
  // ---- enc2: h = relu(h1 @ enc_w2 + b2) ----
  {
    v16h a0 = ldsA(Bt, lane, 0);
    v16h a1 = ldsA(Bt, lane, 32);
#pragma unroll
    for (int nt = 0; nt < 4; ++nt) {
      v8f c = {};
      c = WMMA(a0, glbB(ws + WS_ENC2, lane, 64, nt, 0), c);
      c = WMMA(a1, glbB(ws + WS_ENC2, lane, 64, nt, 1), c);
      storeReluTile(A, lane, nt, enc_b2, c);
    }
  }
  v16h h0 = ldsA(A, lane, 0);
  v16h h1 = ldsA(A, lane, 32);
  const int n = lane & 15, mb = (lane >> 4) * 8;

  // ---- four gates: sigmoid(relu(h@w1+b1)@w2+b2) ----
  const float* gb1s[4] = {td_b1, cd_b1, tu_b1, cu_b1};
  const float* gb2s[4] = {td_b2, cd_b2, tu_b2, cu_b2};
#pragma unroll
  for (int g = 0; g < 4; ++g) {
    const _Float16* w1 = ws + WS_GW1 + g * 4096;
#pragma unroll
    for (int nt = 0; nt < 4; ++nt) {
      v8f c = {};
      c = WMMA(h0, glbB(w1, lane, 64, nt, 0), c);
      c = WMMA(h1, glbB(w1, lane, 64, nt, 1), c);
      storeReluTile(Bt, lane, nt, gb1s[g], c);
    }
    v16h ga0 = ldsA(Bt, lane, 0);
    v16h ga1 = ldsA(Bt, lane, 32);
    const _Float16* w2 = ws + WS_GW2 + g * 1024;
    v8f c2 = {};
    c2 = WMMA(ga0, glbB(w2, lane, 64, 0, 0), c2);
    c2 = WMMA(ga1, glbB(w2, lane, 64, 0, 1), c2);
    if (n < 4) {
      float bs = gb2s[g][n];
#pragma unroll
      for (int v = 0; v < 8; ++v) {
        float xv = c2[v] + bs;
        stok[w * 16 + mb + v][g * 4 + n] = 1.f / (1.f + __expf(-xv));
      }
    }
  }
  // ---- fused Sd | Su | skip projection (64x12 packed into 64x16) ----
  {
    v8f c = {};
    c = WMMA(h0, glbB(ws + WS_SDS, lane, 64, 0, 0), c);
    c = WMMA(h1, glbB(ws + WS_SDS, lane, 64, 0, 1), c);
    if (n < 12) {
      float bs = (n < 4) ? sd_b[n] : (n < 8) ? su_b[n - 4] : skip_b[n - 8];
#pragma unroll
      for (int v = 0; v < 8; ++v)
        stok[w * 16 + mb + v][16 + n] = c[v] + bs;
    }
  }
  __syncthreads();

  const int t = threadIdx.x;
  // ---- down scan: 64 threads, each a (batch,O) pair ----
  if (t < 64) {
    int bb = t >> 2, o = t & 3;
    float d = 0.f, u = 0.f;
#pragma unroll
    for (int l = 0; l < 10; ++l) {
      float* s = stok[l * 16 + bb];
      float dn = s[o] * d + s[4 + o] * u + s[16 + o];
      float un = s[8 + o] * u + s[12 + o] * d + s[20 + o];
      s[28 + o] = dn;
      d = dn; u = un;
    }
  }
  __syncthreads();
  // ---- boundary condition + up scan ----
  if (t < 64) {
    int bb = t >> 2, o = t & 3;
    float* sl = stok[144 + bb];
    float D0 = sl[28], D1 = sl[29], D2 = sl[30], D3 = sl[31];
    float u = bc_b2[o];
    for (int j = 0; j < 32; ++j) {
      float hv = bc_b1[j] + D0 * bc_w1[j] + D1 * bc_w1[32 + j] +
                 D2 * bc_w1[64 + j] + D3 * bc_w1[96 + j];
      hv = fmaxf(hv, 0.f);
      u += hv * bc_w2[j * 4 + o];
    }
#pragma unroll
    for (int l = 9; l >= 0; --l) {
      float* s = stok[l * 16 + bb];
      float un = s[8 + o] * u + s[12 + o] * s[28 + o] + s[20 + o];
      s[32 + o] = un;
      u = un;
    }
  }
  __syncthreads();
  // ---- output heads: 160 threads, each one token ----
  if (t < 160) {
    int bb = t & 15, l = t >> 4;
    float* s = stok[l * 16 + bb];
    float acc[4];
#pragma unroll
    for (int o = 0; o < 4; ++o) acc[o] = s[24 + o] + hd_b2[o] + hu_b2[o];
    for (int j = 0; j < 32; ++j) {
      float hdv = hd_b1[j] + s[28] * hd_w1[j] + s[29] * hd_w1[32 + j] +
                  s[30] * hd_w1[64 + j] + s[31] * hd_w1[96 + j];
      hdv = fmaxf(hdv, 0.f);
      float huv = hu_b1[j] + s[32] * hu_w1[j] + s[33] * hu_w1[32 + j] +
                  s[34] * hu_w1[64 + j] + s[35] * hu_w1[96 + j];
      huv = fmaxf(huv, 0.f);
#pragma unroll
      for (int o = 0; o < 4; ++o)
        acc[o] += hdv * hd_w2[j * 4 + o] + huv * hu_w2[j * 4 + o];
    }
    int b = b0 + bb;
    if (b < Bn) {
#pragma unroll
      for (int o = 0; o < 4; ++o)
        out[((size_t)b * 4 + o) * 10 + l] = acc[o];
    }
  }
}

extern "C" void kernel_launch(void* const* d_in, const int* in_sizes, int n_in,
                              void* d_out, int out_size, void* d_ws, size_t ws_size,
                              hipStream_t stream) {
  const float* x      = (const float*)d_in[0];
  const float* emb    = (const float*)d_in[1];
  const float* enc_w1 = (const float*)d_in[2];
  const float* enc_b1 = (const float*)d_in[3];
  const float* enc_w2 = (const float*)d_in[4];
  const float* enc_b2 = (const float*)d_in[5];
  const float* td_w1 = (const float*)d_in[6];  const float* td_b1 = (const float*)d_in[7];
  const float* td_w2 = (const float*)d_in[8];  const float* td_b2 = (const float*)d_in[9];
  const float* cd_w1 = (const float*)d_in[10]; const float* cd_b1 = (const float*)d_in[11];
  const float* cd_w2 = (const float*)d_in[12]; const float* cd_b2 = (const float*)d_in[13];
  const float* tu_w1 = (const float*)d_in[14]; const float* tu_b1 = (const float*)d_in[15];
  const float* tu_w2 = (const float*)d_in[16]; const float* tu_b2 = (const float*)d_in[17];
  const float* cu_w1 = (const float*)d_in[18]; const float* cu_b1 = (const float*)d_in[19];
  const float* cu_w2 = (const float*)d_in[20]; const float* cu_b2 = (const float*)d_in[21];
  const float* sd_w  = (const float*)d_in[22]; const float* sd_b  = (const float*)d_in[23];
  const float* su_w  = (const float*)d_in[24]; const float* su_b  = (const float*)d_in[25];
  const float* bc_w1 = (const float*)d_in[26]; const float* bc_b1 = (const float*)d_in[27];
  const float* bc_w2 = (const float*)d_in[28]; const float* bc_b2 = (const float*)d_in[29];
  const float* skip_w = (const float*)d_in[30]; const float* skip_b = (const float*)d_in[31];
  const float* hd_w1 = (const float*)d_in[32]; const float* hd_b1 = (const float*)d_in[33];
  const float* hd_w2 = (const float*)d_in[34]; const float* hd_b2 = (const float*)d_in[35];
  const float* hu_w1 = (const float*)d_in[36]; const float* hu_b1 = (const float*)d_in[37];
  const float* hu_w2 = (const float*)d_in[38]; const float* hu_b2 = (const float*)d_in[39];

  _Float16* ws = (_Float16*)d_ws;
  float* out = (float*)d_out;
  int Bn = in_sizes[0] / 60;               // x is (B, 6, 10)

  prepack_kernel<<<64, 256, 0, stream>>>(enc_w1, enc_w2,
      td_w1, cd_w1, tu_w1, cu_w1, td_w2, cd_w2, tu_w2, cu_w2,
      sd_w, su_w, skip_w, ws);

  int grid = (Bn + 15) / 16;
  vrtcn_fused<<<grid, 320, 0, stream>>>(x, emb, enc_b1, enc_b2,
      td_b1, td_b2, cd_b1, cd_b2, tu_b1, tu_b2, cu_b1, cu_b2,
      sd_b, su_b, skip_b, bc_w1, bc_b1, bc_w2, bc_b2,
      hd_w1, hd_b1, hd_w2, hd_b2, hu_w1, hu_b1, hu_w2, hu_b2,
      ws, out, Bn);
}
